// LaneATT_62362925138765
// MI455X (gfx1250) — matrix operand order repaired and checked by reading, not compiled, not run on hardware
//
#include <hip/hip_runtime.h>
#include <hip/hip_bf16.h>

// ---------------- problem constants ----------------
#define B_    32
#define CB_   512
#define FH_   11
#define FW_   20
#define HW_   (FH_*FW_)      // 220
#define CAF_  64
#define N_    1000
#define S_    72
#define D_    (CAF_*FH_)     // 704
#define D2_   (2*D_)         // 1408
#define NM1_  (N_-1)         // 999
#define NO_   (S_+5)         // 77 output width
#define CR_   (S_+3)         // 75 cls+reg rows
#define SLD_  1024           // scores row pitch (f32)
#define ALD_  1024           // att_mat K pitch / padded rows, bafT N pitch
#define CRLD_ 80             // cls/reg C pitch

typedef __attribute__((ext_vector_type(16))) __bf16 v16bf;
typedef __attribute__((ext_vector_type(8)))  float  v8f;
typedef __hip_bfloat16 bf16_t;

struct FragBF { union { uint4 q[2]; v16bf v; }; };

// ---------------- 1x1 conv: feat[b,o,h,w] = sum_c x[b,c,h,w]*w1[o,c] + b1[o] ----
__global__ __launch_bounds__(256) void conv1x1_kernel(
    const float* __restrict__ x, const float* __restrict__ w1,
    const float* __restrict__ b1, float* __restrict__ feat)
{
  int bo = blockIdx.x;
  int b  = bo >> 6;          // /CAF_
  int o  = bo & 63;
  int hw = threadIdx.x;
  if (hw >= HW_) return;
  const float* xp = x + ((size_t)b * CB_) * HW_ + hw;
  const float* wp = w1 + (size_t)o * CB_;
  float acc = b1[o];
#pragma unroll 8
  for (int c = 0; c < CB_; ++c) acc = fmaf(xp[(size_t)c * HW_], wp[c], acc);
  feat[((size_t)b * CAF_ + o) * HW_ + hw] = acc;
}

// ---------------- gather + mask -> baf (bf16, into all_feat right half) + bafT ----
__global__ __launch_bounds__(256) void gather_kernel(
    const float* __restrict__ feat, const int* __restrict__ cut_xs,
    const unsigned char* __restrict__ invalid,
    bf16_t* __restrict__ allfeat, bf16_t* __restrict__ bafT)
{
  long idx = (long)blockIdx.x * blockDim.x + threadIdx.x;
  if (idx >= (long)B_ * N_ * D_) return;
  int  d  = (int)(idx % D_);
  long bn = idx / D_;
  int  n  = (int)(bn % N_);
  int  b  = (int)(bn / N_);
  int  c  = d / FH_;
  int  h  = d - c * FH_;
  int  xc = cut_xs[(size_t)n * D_ + d];
  float v = 0.0f;
  if (!invalid[n * FH_ + h])
    v = feat[(((size_t)b * CAF_ + c) * FH_ + h) * FW_ + xc];
  bf16_t hv = __float2bfloat16(v);
  allfeat[(size_t)bn * D2_ + D_ + d] = hv;                 // baf half of all_feat
  bafT[((size_t)b * D_ + d) * ALD_ + n] = hv;              // transposed copy (K-contig)
}

// zero the bafT pad columns [N_, ALD_) so WMMA never sees uninitialized NaNs
__global__ void padfill_bafT(bf16_t* __restrict__ bafT)
{
  int idx = blockIdx.x * blockDim.x + threadIdx.x;
  const int padw = ALD_ - N_;
  if (idx >= B_ * D_ * padw) return;
  int j  = N_ + idx % padw;
  int bd = idx / padw;
  bafT[(size_t)bd * ALD_ + j] = __float2bfloat16(0.0f);
}

// ---------------- weight conversions ----------------
__global__ void cvt_attnw(const float* __restrict__ w, bf16_t* __restrict__ wh)
{
  int idx = blockIdx.x * blockDim.x + threadIdx.x;
  if (idx < NM1_ * D_) wh[idx] = __float2bfloat16(w[idx]);
}

__global__ void cvt_crw(const float* __restrict__ cls_w, const float* __restrict__ reg_w,
                        bf16_t* __restrict__ crw)
{
  int idx = blockIdx.x * blockDim.x + threadIdx.x;
  if (idx >= CR_ * D2_) return;
  int o = idx / D2_, k = idx - o * D2_;
  float v = (o < 2) ? cls_w[(size_t)o * D2_ + k] : reg_w[(size_t)(o - 2) * D2_ + k];
  crw[idx] = __float2bfloat16(v);
}

// ---------------- generic WMMA GEMM: C = A * B^T  (bf16 in, f32 acc) ------------
// A: (M x K) row-major pitch lda ; B: (Nn x K) row-major pitch ldb (i.e. B^T cols)
// wave tile: 32(M) x 64(N) = 8 accumulators; block = 8 waves = 256 M rows.
// M rows must be readable up to the 256-row block boundary (buffers padded);
// out-of-range N lanes are CLAMPED to row Nn-1 (loop-invariant, no EXEC churn) —
// their garbage accumulator columns are dropped by the store guard.
__global__ __launch_bounds__(256) void gemm_bf16_kernel(
    const bf16_t* __restrict__ A, long lda, long sA,
    const bf16_t* __restrict__ B, long ldb, long sB,
    void* __restrict__ Cp, long ldc, long sC,
    int M, int Nn, int K, int c_is_bf16)
{
  const int wave = threadIdx.x >> 5;
  const int lane = threadIdx.x & 31;
  const int hi   = lane >> 4;        // half-wave selector (matches ISA VGPR layout)
  const int lm   = lane & 15;
  const int m0   = (blockIdx.y * 8 + wave) * 32;
  const int n0   = blockIdx.x * 64;
  const long bz  = blockIdx.z;

  A += bz * sA;
  B += bz * sB;
  char* Cb = (char*)Cp + bz * sC * (c_is_bf16 ? 2 : 4);

  v8f acc[2][4];
#pragma unroll
  for (int t = 0; t < 2; ++t)
#pragma unroll
    for (int s = 0; s < 4; ++s) acc[t][s] = (v8f){};

  // A fragment pointers: lane holds A[row, k0+8*hi .. +7] and [.. +16 .. +23]
  const bf16_t* ap0 = A + (long)(m0 + lm) * lda + 8 * hi;
  const bf16_t* ap1 = ap0 + 16 * lda;

  // B fragment pointers, row clamped once (loop-invariant guard hoist)
  const bf16_t* bp[4];
#pragma unroll
  for (int s = 0; s < 4; ++s) {
    int n = n0 + 16 * s + lm;
    if (n >= Nn) n = Nn - 1;
    bp[s] = B + (long)n * ldb + 8 * hi;
  }

#pragma unroll 2
  for (int k0 = 0; k0 < K; k0 += 32) {
    FragBF a0, a1, bfr[4];
    const uint4* p;
    p = (const uint4*)(ap0 + k0); a0.q[0] = p[0]; a0.q[1] = p[2];
    p = (const uint4*)(ap1 + k0); a1.q[0] = p[0]; a1.q[1] = p[2];
#pragma unroll
    for (int s = 0; s < 4; ++s) {
      p = (const uint4*)(bp[s] + k0);
      bfr[s].q[0] = p[0];
      bfr[s].q[1] = p[2];
    }
    if (k0 + 32 < K) {
      __builtin_prefetch((const void*)(ap0 + k0 + 32), 0, 0);  // global_prefetch_b8
      __builtin_prefetch((const void*)(ap1 + k0 + 32), 0, 0);
    }
#pragma unroll
    for (int s = 0; s < 4; ++s) {
      acc[0][s] = __builtin_amdgcn_wmma_f32_16x16x32_bf16(false, a0.v, false, bfr[s].v,
                                                          (short)0, acc[0][s], false, false);
      acc[1][s] = __builtin_amdgcn_wmma_f32_16x16x32_bf16(false, a1.v, false, bfr[s].v,
                                                          (short)0, acc[1][s], false, false);
    }
  }

#pragma unroll
  for (int t = 0; t < 2; ++t) {
#pragma unroll
    for (int s = 0; s < 4; ++s) {
      const int n = n0 + 16 * s + lm;
      if (n >= Nn) continue;
#pragma unroll
      for (int r = 0; r < 8; ++r) {
        const int m = m0 + 16 * t + r + 8 * hi;  // C layout: VGPR r, lanes>=16 -> M+8
        if (m < M) {
          if (c_is_bf16)
            ((bf16_t*)Cb)[(long)m * ldc + n] = __float2bfloat16(acc[t][s][r]);
          else
            ((float*)Cb)[(long)m * ldc + n] = acc[t][s][r];
        }
      }
    }
  }
}

// ---------------- softmax over 999 + zero-diagonal scatter to bf16 att_mat row ----
__global__ __launch_bounds__(256) void softmax_diag_kernel(
    const float* __restrict__ scores, const float* __restrict__ attn_b,
    bf16_t* __restrict__ attm)
{
  const int row = blockIdx.x;              // b*N_ + i
  const int b   = row / N_;
  const int i   = row - b * N_;
  const float* s = scores + (size_t)row * SLD_;
  bf16_t* arow = attm + ((size_t)b * ALD_ + i) * ALD_;

  __shared__ float vals[NM1_ + 1];
  __shared__ float red[256];
  const int t = threadIdx.x;

  float mx = -3.4e38f;
  for (int m = t; m < NM1_; m += 256) {
    float v = s[m] + attn_b[m];
    vals[m] = v;
    mx = fmaxf(mx, v);
  }
  red[t] = mx; __syncthreads();
  for (int off = 128; off > 0; off >>= 1) {
    if (t < off) red[t] = fmaxf(red[t], red[t + off]);
    __syncthreads();
  }
  mx = red[0]; __syncthreads();

  float sum = 0.0f;
  for (int m = t; m < NM1_; m += 256) {
    float e = __expf(vals[m] - mx);
    vals[m] = e;
    sum += e;
  }
  red[t] = sum; __syncthreads();
  for (int off = 128; off > 0; off >>= 1) {
    if (t < off) red[t] += red[t + off];
    __syncthreads();
  }
  const float inv = 1.0f / red[0];
  __syncthreads();

  // att_mat row i: j==i -> 0 ; j<i -> att[j] ; j>i -> att[j-1] ; pad cols -> 0
  for (int j = t; j < ALD_; j += 256) {
    float v = 0.0f;
    if (j < N_ && j != i) v = vals[j - (j > i ? 1 : 0)] * inv;
    arow[j] = __float2bfloat16(v);
  }
}

// ---------------- final assembly: (B,N,77) ----------------
__global__ __launch_bounds__(256) void assemble_kernel(
    const float* __restrict__ crout, const float* __restrict__ anchors,
    const float* __restrict__ cls_b, const float* __restrict__ reg_b,
    float* __restrict__ out)
{
  long idx = (long)blockIdx.x * blockDim.x + threadIdx.x;
  if (idx >= (long)B_ * N_ * NO_) return;
  int  o   = (int)(idx % NO_);
  long row = idx / NO_;                    // b*N_ + n
  int  n   = (int)(row % N_);
  float v;
  if (o < 2)      v = crout[row * CRLD_ + o] + cls_b[o];
  else if (o < 4) v = anchors[(size_t)n * NO_ + o];
  else            v = anchors[(size_t)n * NO_ + o] + crout[row * CRLD_ + (o - 2)] + reg_b[o - 4];
  out[idx] = v;
}

// ---------------- host launcher ----------------
extern "C" void kernel_launch(void* const* d_in, const int* in_sizes, int n_in,
                              void* d_out, int out_size, void* d_ws, size_t ws_size,
                              hipStream_t stream)
{
  (void)in_sizes; (void)n_in; (void)out_size; (void)ws_size;
  const float* x        = (const float*)d_in[0];
  const float* anchors  = (const float*)d_in[1];
  const float* conv1_w  = (const float*)d_in[2];
  const float* conv1_b  = (const float*)d_in[3];
  const float* attn_w   = (const float*)d_in[4];
  const float* attn_b   = (const float*)d_in[5];
  const float* cls_w    = (const float*)d_in[6];
  const float* cls_b    = (const float*)d_in[7];
  const float* reg_w    = (const float*)d_in[8];
  const float* reg_b    = (const float*)d_in[9];
  const int*   cut_xs   = (const int*)d_in[12];
  const unsigned char* invalid = (const unsigned char*)d_in[13];
  float* out = (float*)d_out;

  // workspace carve-out (256B aligned)
  size_t off = 0;
  auto carve = [&](size_t bytes) -> char* {
    char* p = (char*)d_ws + off;
    off = (off + bytes + 255) & ~(size_t)255;
    return p;
  };
  float*  feat    = (float*) carve((size_t)B_ * CAF_ * HW_ * 4);          // 1.8 MB
  float*  scores  = (float*) carve((size_t)B_ * N_ * SLD_ * 4);           // 131 MB
  bf16_t* attm    = (bf16_t*)carve((size_t)B_ * ALD_ * ALD_ * 2);         // 67 MB (rows padded)
  bf16_t* allfeat = (bf16_t*)carve((size_t)B_ * N_ * D2_ * 2);            // 90 MB
  bf16_t* bafT    = (bf16_t*)carve((size_t)B_ * D_ * ALD_ * 2);           // 46 MB
  float*  crout   = (float*) carve((size_t)B_ * N_ * CRLD_ * 4);          // 10 MB
  bf16_t* attnw_h = (bf16_t*)carve((size_t)NM1_ * D_ * 2);
  bf16_t* crw_h   = (bf16_t*)carve((size_t)CR_ * D2_ * 2);

  // weight conversions
  cvt_attnw<<<(NM1_ * D_ + 255) / 256, 256, 0, stream>>>(attn_w, attnw_h);
  cvt_crw  <<<(CR_ * D2_ + 255) / 256, 256, 0, stream>>>(cls_w, reg_w, crw_h);

  // conv + gather
  conv1x1_kernel<<<B_ * CAF_, 256, 0, stream>>>(x, conv1_w, conv1_b, feat);
  {
    long tot = (long)B_ * N_ * D_;
    gather_kernel<<<(unsigned)((tot + 255) / 256), 256, 0, stream>>>(
        feat, cut_xs, invalid, allfeat, bafT);
  }
  padfill_bafT<<<(B_ * D_ * (ALD_ - N_) + 255) / 256, 256, 0, stream>>>(bafT);

  // scores = baf (32000 x 704) * attn_w^T (704 x 999); 256 rows / block
  gemm_bf16_kernel<<<dim3((NM1_ + 63) / 64, (B_ * N_) / 256, 1), 256, 0, stream>>>(
      allfeat + D_, (long)D2_, 0L,
      attnw_h, (long)D_, 0L,
      (void*)scores, (long)SLD_, 0L,
      B_ * N_, NM1_, D_, 0);

  // softmax + diagonal scatter -> bf16 att_mat (K padded to 1024 with zeros)
  softmax_diag_kernel<<<B_ * N_, 256, 0, stream>>>(scores, attn_b, attm);

  // att_feat[b] = att_mat[b] (1000 x 1024) * bafT[b]^T (1024 x 704) -> left half of all_feat
  // grid.y = 4 -> 1024 rows, exactly the padded att_mat row count
  gemm_bf16_kernel<<<dim3(D_ / 64, (ALD_) / 256, B_), 256, 0, stream>>>(
      attm, (long)ALD_, (long)ALD_ * ALD_,
      bafT, (long)ALD_, (long)D_ * ALD_,
      (void*)allfeat, (long)D2_, (long)N_ * D2_,
      N_, D_, ALD_, 1);

  // cls/reg = all_feat (32000 x 1408) * crw^T (1408 x 75)
  gemm_bf16_kernel<<<dim3((CR_ + 63) / 64, (B_ * N_) / 256, 1), 256, 0, stream>>>(
      allfeat, (long)D2_, 0L,
      crw_h, (long)D2_, 0L,
      (void*)crout, (long)CRLD_, 0L,
      B_ * N_, CR_, D2_, 0);

  // final output assembly
  {
    long tot = (long)B_ * N_ * NO_;
    assemble_kernel<<<(unsigned)((tot + 255) / 256), 256, 0, stream>>>(
        crout, anchors, cls_b, reg_b, out);
  }
}
